// AnomalyMoE_24060406792290
// MI455X (gfx1250) — matrix operand order repaired and checked by reading, compiled, not run on hardware
//
#include <hip/hip_runtime.h>
#include <hip/hip_bf16.h>
#include <math.h>

// ---------------------------------------------------------------------------
// AnomalyMoE kNN cosine-similarity heatmap for MI455X (gfx1250, wave32, WMMA)
//   L=4 layers, N=256 query patches, M=51200 gallery patches, D=1024
//   out[n] = 1 - mean_l( max_m  cos(q[l,n], g[l,m]) )   -> 256 floats
//
// Pipeline: fp32 -> L2-normalized f16 (one streaming pass), then a WMMA GEMM
// with online max: gallery tiles staged in LDS via async global->LDS copies
// (ASYNCcnt double buffering), consumed by 8 waves x 2 query tiles each.
// ---------------------------------------------------------------------------

#define EMB      1024
#define NQ       256
#define NGAL     51200
#define NLAYERS  4
#define MTILES   (NGAL / 16)        // 3200 gallery tiles of 16 rows

// LDS staging: 16 rows x 1024 f16, each row padded by 8 halfs (16B) so the
// 16-lane ds_load_b128 B-fragment reads land on distinct bank groups.
#define RS_HALF  (EMB + 8)          // 1032 halfs per row
#define RS_BYTE  (RS_HALF * 2)      // 2064 bytes per row
#define BUFBYTE  (16 * RS_BYTE)     // 33024 bytes per tile buffer
#define SMEMBYTE (2 * BUFBYTE)      // double buffered

typedef _Float16 v16h __attribute__((ext_vector_type(16)));
typedef _Float16 v8h  __attribute__((ext_vector_type(8)));
typedef _Float16 v4h  __attribute__((ext_vector_type(4)));
typedef float    v8f  __attribute__((ext_vector_type(8)));
typedef int      gv4i __attribute__((vector_size(16)));   // matches builtin proto

#if __has_builtin(__builtin_amdgcn_global_load_async_to_lds_b128)
#define HAS_ASYNC_LDS 1
#else
#define HAS_ASYNC_LDS 0
#endif

// order-preserving float <-> uint map so we can use integer atomicMax
__device__ __forceinline__ unsigned f2ord(float f) {
    unsigned u = __float_as_uint(f);
    return (u & 0x80000000u) ? ~u : (u | 0x80000000u);
}
__device__ __forceinline__ float ord2f(unsigned u) {
    return (u & 0x80000000u) ? __uint_as_float(u & 0x7FFFFFFFu)
                             : __uint_as_float(~u);
}

// one 16-byte global -> LDS transfer (async on gfx1250, sync fallback)
__device__ __forceinline__ void copy16_g2l(const _Float16* g, _Float16* l) {
#if HAS_ASYNC_LDS
    __builtin_amdgcn_global_load_async_to_lds_b128((gv4i*)g, (gv4i*)l, 0, 0);
#else
    *(uint4*)l = *(const uint4*)g;
#endif
}
__device__ __forceinline__ void wait_async_le8() {
#if HAS_ASYNC_LDS
#if __has_builtin(__builtin_amdgcn_s_wait_asynccnt)
    __builtin_amdgcn_s_wait_asynccnt(8);
#else
    asm volatile("s_wait_asynccnt 0x8" ::: "memory");
#endif
#endif
}
__device__ __forceinline__ void wait_async_le0() {
#if HAS_ASYNC_LDS
#if __has_builtin(__builtin_amdgcn_s_wait_asynccnt)
    __builtin_amdgcn_s_wait_asynccnt(0);
#else
    asm volatile("s_wait_asynccnt 0x0" ::: "memory");
#endif
#endif
}

// ---------------------------------------------------------------------------
// Kernel 1: L2-normalize fp32 rows of length 1024 and emit f16.
// One wave (32 lanes) per row; lane handles 32 strided elements.
// ---------------------------------------------------------------------------
__global__ __launch_bounds__(256) void normalize_f16_kernel(
    const float* __restrict__ in, _Float16* __restrict__ out, int rows)
{
    const int lane = threadIdx.x & 31;
    const int row  = blockIdx.x * 8 + (threadIdx.x >> 5);
    if (row >= rows) return;

    const float* src = in + (size_t)row * EMB;
    float4 v[8];
    float  ss = 0.0f;
#pragma unroll
    for (int i = 0; i < 8; ++i) {
        v[i] = *(const float4*)(src + lane * 4 + i * 128);
        ss += v[i].x * v[i].x + v[i].y * v[i].y + v[i].z * v[i].z + v[i].w * v[i].w;
    }
#pragma unroll
    for (int m = 1; m <= 16; m <<= 1) ss += __shfl_xor(ss, m, 32);

    const float scale = 1.0f / fmaxf(sqrtf(ss), 1e-8f);

    _Float16* dst = out + (size_t)row * EMB;
#pragma unroll
    for (int i = 0; i < 8; ++i) {
        v4h h = { (_Float16)(v[i].x * scale), (_Float16)(v[i].y * scale),
                  (_Float16)(v[i].z * scale), (_Float16)(v[i].w * scale) };
        *(v4h*)(dst + lane * 4 + i * 128) = h;
    }
}

// ---------------------------------------------------------------------------
// Kernel 2: init the atomic-max key buffer (4*256 uints) to ordered(-inf)=0.
// ---------------------------------------------------------------------------
__global__ void init_keys_kernel(unsigned* __restrict__ keys)
{
    keys[blockIdx.x * blockDim.x + threadIdx.x] = 0u;
}

// ---------------------------------------------------------------------------
// Kernel 3: WMMA GEMM + online max over gallery.
//   grid.y = 4  : one layer per y; block covers all 16 query tiles
//   grid.x      : gallery chunks; block loops over its share of 3200 M-tiles
//   8 waves/block, each wave owns query tiles {w, w+8} (register blocking)
//   Gallery tile (16x1024 f16 = 32KB) double-buffered in LDS via async copies.
// ---------------------------------------------------------------------------
__global__ __launch_bounds__(256) void wmma_simmax_kernel(
    const _Float16* __restrict__ qn,   // [4,256,1024]  f16, L2-normalized
    const _Float16* __restrict__ gn,   // [4,51200,1024] f16, L2-normalized
    unsigned* __restrict__ keys)       // [4,256] ordered-uint max keys
{
    extern __shared__ __align__(16) char smem[];

    const int tid      = threadIdx.x;
    const int lane     = tid & 31;
    const int lane16   = lane & 15;
    const int halfSel  = lane >> 4;              // 0: lanes 0-15, 1: lanes 16-31
    const int wave     = tid >> 5;               // 0..7
    const int layer    = (int)blockIdx.y;        // 0..3
    const int qtile0   = wave;                   // first query tile of this wave
    const int qtile1   = wave + 8;               // second query tile

    const _Float16* gLayer = gn + (size_t)layer * NGAL * EMB;

    // A-fragment rows (contiguous k order, same as B; not the ISA table order):
    //   lanes 0-15: M=lane, K = k0+0..15 contiguous
    //   lanes 16-31: M=lane-16, K = k0+16..31 contiguous
    const _Float16* aRow0 =
        qn + ((size_t)layer * NQ + (size_t)qtile0 * 16 + lane16) * EMB;
    const _Float16* aRow1 =
        qn + ((size_t)layer * NQ + (size_t)qtile1 * 16 + lane16) * EMB;
    const int aOff = halfSel * 16;
    // B-fragment (32x16, row-striped): lanes 0-15 -> K=k0+0..15 contiguous,
    // lanes 16-31 -> K=k0+16..31 contiguous, N = lane16 = gallery row in tile.
    const int bOff = halfSel * 16;

    v8f maxv0, maxv1;
#pragma unroll
    for (int r = 0; r < 8; ++r) { maxv0[r] = -2.0f; maxv1[r] = -2.0f; }

    const int mt0    = (int)blockIdx.x;
    const int stride = (int)gridDim.x;

    // ---- issue first tile into buffer 0 ----
    {
        const _Float16* gt = gLayer + (size_t)mt0 * 16 * EMB;
#pragma unroll
        for (int i = 0; i < 8; ++i) {
            const int c   = tid + i * 256;        // chunk id 0..2047
            const int row = c >> 7;               // 0..15
            const int col = c & 127;              // 16B chunk within row
            copy16_g2l(gt + (size_t)row * EMB + col * 8,
                       (_Float16*)(smem + (size_t)row * RS_BYTE) + col * 8);
        }
    }

    int it = 0;
    for (int mt = mt0; mt < MTILES; mt += stride, ++it) {
        const int  buf  = it & 1;
        const int  nmt  = mt + stride;

        if (nmt < MTILES) {
            // issue next tile into the other buffer, then wait for current
            const _Float16* gt = gLayer + (size_t)nmt * 16 * EMB;
            char* lbuf = smem + (size_t)(buf ^ 1) * BUFBYTE;
#pragma unroll
            for (int i = 0; i < 8; ++i) {
                const int c   = tid + i * 256;
                const int row = c >> 7;
                const int col = c & 127;
                copy16_g2l(gt + (size_t)row * EMB + col * 8,
                           (_Float16*)(lbuf + (size_t)row * RS_BYTE) + col * 8);
            }
            wait_async_le8();
        } else {
            wait_async_le0();
        }
        __syncthreads();   // current tile fully resident in LDS for all waves

        const _Float16* bRow =
            (const _Float16*)(smem + (size_t)buf * BUFBYTE) +
            (size_t)lane16 * RS_HALF;

        v8f acc0 = {}, acc1 = {};
#pragma unroll 4
        for (int k0 = 0; k0 < EMB; k0 += 32) {
            v16h b = *(const v16h*)(bRow + k0 + bOff);           // LDS (ds_load)
            v8h a0l = *(const v8h*)(aRow0 + k0 + aOff);
            v8h a0h = *(const v8h*)(aRow0 + k0 + 8 + aOff);
            v8h a1l = *(const v8h*)(aRow1 + k0 + aOff);
            v8h a1h = *(const v8h*)(aRow1 + k0 + 8 + aOff);
            v16h a0 = __builtin_shufflevector(a0l, a0h,
                        0, 1, 2, 3, 4, 5, 6, 7, 8, 9, 10, 11, 12, 13, 14, 15);
            v16h a1 = __builtin_shufflevector(a1l, a1h,
                        0, 1, 2, 3, 4, 5, 6, 7, 8, 9, 10, 11, 12, 13, 14, 15);
            acc0 = __builtin_amdgcn_wmma_f32_16x16x32_f16(
                       false, a0, false, b, (short)0, acc0, false, false);
            acc1 = __builtin_amdgcn_wmma_f32_16x16x32_f16(
                       false, a1, false, b, (short)0, acc1, false, false);
        }
#pragma unroll
        for (int r = 0; r < 8; ++r) {
            maxv0[r] = fmaxf(maxv0[r], acc0[r]);
            maxv1[r] = fmaxf(maxv1[r], acc1[r]);
        }
        __syncthreads();   // all waves done reading buf before it is refilled
    }

    // max over the 16 gallery columns held by each half-wave
#pragma unroll
    for (int m = 1; m <= 8; m <<= 1) {
#pragma unroll
        for (int r = 0; r < 8; ++r) {
            maxv0[r] = fmaxf(maxv0[r], __shfl_xor(maxv0[r], m, 32));
            maxv1[r] = fmaxf(maxv1[r], __shfl_xor(maxv1[r], m, 32));
        }
    }

    // C/D layout: VGPR r -> query row (r + 8*halfSel); publish via atomic max
    if (lane16 == 0) {
#pragma unroll
        for (int r = 0; r < 8; ++r) {
            const int q0 = qtile0 * 16 + halfSel * 8 + r;
            const int q1 = qtile1 * 16 + halfSel * 8 + r;
            atomicMax(keys + layer * NQ + q0, f2ord(maxv0[r]));
            atomicMax(keys + layer * NQ + q1, f2ord(maxv1[r]));
        }
    }
}

// ---------------------------------------------------------------------------
// Kernel 4: mean over layers, anomaly map = 1 - mean.
// ---------------------------------------------------------------------------
__global__ __launch_bounds__(256) void finalize_kernel(
    const unsigned* __restrict__ keys, float* __restrict__ out)
{
    const int n = threadIdx.x;       // 256
    float s = 0.0f;
#pragma unroll
    for (int l = 0; l < NLAYERS; ++l) s += ord2f(keys[l * NQ + n]);
    out[n] = 1.0f - s * (1.0f / NLAYERS);
}

// ---------------------------------------------------------------------------
extern "C" void kernel_launch(void* const* d_in, const int* in_sizes, int n_in,
                              void* d_out, int out_size, void* d_ws, size_t ws_size,
                              hipStream_t stream)
{
    const float* test = (const float*)d_in[0];   // [4,256,1024]   fp32
    const float* gal  = (const float*)d_in[1];   // [4,51200,1024] fp32
    float*       out  = (float*)d_out;           // 256 fp32

    const size_t QROWS  = (size_t)NLAYERS * NQ;                 // 1024
    const size_t GROWS  = (size_t)NLAYERS * NGAL;               // 204800
    const size_t QBYTES = QROWS * EMB * sizeof(_Float16);       // 2 MB
    const size_t GBYTES = GROWS * EMB * sizeof(_Float16);       // 400 MB

    char*      ws   = (char*)d_ws;
    _Float16*  qn   = (_Float16*)ws;
    _Float16*  gn   = (_Float16*)(ws + QBYTES);
    unsigned*  keys = (unsigned*)(ws + QBYTES + GBYTES);        // 4 KB

    // 1) normalize queries + gallery into f16 (one wave per row)
    normalize_f16_kernel<<<dim3((unsigned)((QROWS + 7) / 8)), 256, 0, stream>>>(
        test, qn, (int)QROWS);
    normalize_f16_kernel<<<dim3((unsigned)((GROWS + 7) / 8)), 256, 0, stream>>>(
        gal, gn, (int)GROWS);

    // 2) reset atomic max keys
    init_keys_kernel<<<dim3(NLAYERS), NQ, 0, stream>>>(keys);

    // 3) WMMA similarity + online gallery max (LDS double-buffered gallery)
    wmma_simmax_kernel<<<dim3(400, NLAYERS), 256, SMEMBYTE, stream>>>(qn, gn, keys);

    // 4) mean over layers, 1 - x
    finalize_kernel<<<1, NQ, 0, stream>>>(keys, out);
}
